// MultiHeadAttention_89086211654132
// MI455X (gfx1250) — compile-verified, matrix operand order checked by reading, not run on hardware
//
#include <hip/hip_runtime.h>

#define S_      2048
#define D_      1024
#define H_      16
#define DEPTH_  64
#define B_      2

typedef __attribute__((ext_vector_type(16))) __bf16 v16bf;
typedef __attribute__((ext_vector_type(8)))  __bf16 v8bf;
typedef __attribute__((ext_vector_type(8)))  float  v8f;

// ---- WMMA helper: D = A(16x32 bf16) * B(32x16 bf16) + C(16x16 f32) ----
static __device__ __forceinline__ v8f wmma_bf16(v16bf a, v16bf b, v8f c) {
  return __builtin_amdgcn_wmma_f32_16x16x32_bf16(
      /*neg_a=*/false, a, /*neg_b=*/false, b,
      /*c_mod=*/(short)0, c, /*reuse_a=*/false, /*reuse_b=*/false);
}

// Fragment element j (lane l) holds K = khalf + (j>>3)*16 + (j&7), khalf = (l>>4)*8.
// With a contiguous-K source, that is two 16B loads at +0 and +16.
static __device__ __forceinline__ v16bf load_frag_bf16(const __bf16* base) {
  v8bf lo = *(const v8bf*)(base);
  v8bf hi = *(const v8bf*)(base + 16);
  return __builtin_shufflevector(lo, hi, 0,1,2,3,4,5,6,7,8,9,10,11,12,13,14,15);
}

static __device__ __forceinline__ v16bf load_frag_f32(const float* base) {
  v16bf a;
#pragma unroll
  for (int g = 0; g < 2; ++g) {
    float4 x = *(const float4*)(base + g*16);
    float4 y = *(const float4*)(base + g*16 + 4);
    a[g*8+0] = (__bf16)x.x; a[g*8+1] = (__bf16)x.y;
    a[g*8+2] = (__bf16)x.z; a[g*8+3] = (__bf16)x.w;
    a[g*8+4] = (__bf16)y.x; a[g*8+5] = (__bf16)y.y;
    a[g*8+6] = (__bf16)y.z; a[g*8+7] = (__bf16)y.w;
  }
  return a;
}

// ---- Weight prep: f32 [K,N] -> bf16 [N,K], LDS-tiled (coalesced both sides) ----
__global__ void w_transpose_bf16_kernel(const float* __restrict__ W,
                                        __bf16* __restrict__ Wt) {
  __shared__ float tile[32][33];
  const int k0 = (blockIdx.x & 31) * 32, n0 = (blockIdx.x >> 5) * 32;
  const int tx = threadIdx.x, ty = threadIdx.y;     // 32 x 8
#pragma unroll
  for (int i = 0; i < 32; i += 8)
    tile[ty + i][tx] = W[(size_t)(k0 + ty + i) * D_ + n0 + tx];
  __syncthreads();
#pragma unroll
  for (int i = 0; i < 32; i += 8)
    Wt[(size_t)(n0 + ty + i) * D_ + k0 + tx] = (__bf16)tile[tx][ty + i];
}

// ---- Projection GEMM: 16x64 strip per wave (A reused across 4 N-tiles) ----
__global__ void qkv_proj_kernel(const float*  __restrict__ X,
                                const __bf16* __restrict__ Wt,
                                const float*  __restrict__ bias,
                                __bf16* __restrict__ Out,
                                int v_transposed) {
  const int lane  = threadIdx.x;
  const int tid   = blockIdx.x * 8 + threadIdx.y;   // 8 waves / block
  const int tileM = tid >> 4;                       // 256 M tiles
  const int tileN = tid & 15;                       // 16 strips of 64 cols
  const int row0 = tileM * 16, col0 = tileN * 64;
  const int l15 = lane & 15, half = lane >> 4, khalf = half * 8;

  const float*  arow = X  + (size_t)(row0 + l15) * D_ + khalf;
  const __bf16* brow = Wt + (size_t)(col0 + l15) * D_ + khalf;

  v8f acc[4] = {{}, {}, {}, {}};
#pragma unroll 2
  for (int k0 = 0; k0 < D_; k0 += 32) {
    __builtin_prefetch(arow + k0 + 128, 0, 1);
    v16bf a = load_frag_f32(arow + k0);
#pragma unroll
    for (int t = 0; t < 4; ++t) {
      v16bf b = load_frag_bf16(brow + (size_t)t * 16 * D_ + k0);
      acc[t] = wmma_bf16(a, b, acc[t]);
    }
  }

#pragma unroll
  for (int t = 0; t < 4; ++t) {
    const int nc = col0 + t * 16 + l15;
    const float bv = bias[nc];
    const int h = nc >> 6, dd = nc & 63;
#pragma unroll
    for (int r = 0; r < 8; ++r) {
      const int rowg = row0 + r + 8 * half;         // row in [B*S]
      const int bb = rowg >> 11, ss = rowg & (S_ - 1);
      const float val = acc[t][r] + bv;
      if (v_transposed)  // V: [B*H, DEPTH, S] so PV B-frags are contiguous in K
        Out[(((size_t)bb * H_ + h) * DEPTH_ + dd) * S_ + ss] = (__bf16)val;
      else               // Q,K: [B*H, S, DEPTH]
        Out[(((size_t)bb * H_ + h) * S_ + ss) * DEPTH_ + dd] = (__bf16)val;
    }
  }
}

// ---- Causal flash attention: 16-query tile per wave, 64 keys / iter ----
__global__ void flash_attn_kernel(const __bf16* __restrict__ Qh,
                                  const __bf16* __restrict__ Kh,
                                  const __bf16* __restrict__ VhT,
                                  __bf16* __restrict__ AttnOut) {
  __shared__ __bf16 lds_p[4][16 * 64];              // per-wave P staging (2 KB each)
  const int lane = threadIdx.x, wave = threadIdx.y;
  const int bh    = blockIdx.x >> 5;                // 32 query-chunks of 64 per (b,h)
  const int chunk = blockIdx.x & 31;
  const int q0 = chunk * 64 + wave * 16;
  const int l15 = lane & 15, half = lane >> 4, khalf = half * 8;

  const __bf16* Qb = Qh  + (size_t)bh * S_ * DEPTH_;
  const __bf16* Kb = Kh  + (size_t)bh * S_ * DEPTH_;
  const __bf16* Vb = VhT + (size_t)bh * DEPTH_ * S_;

  v16bf qa[2];                                      // Q tile 16x64 = 2 A-frags
#pragma unroll
  for (int t = 0; t < 2; ++t)
    qa[t] = load_frag_bf16(Qb + (size_t)(q0 + l15) * DEPTH_ + t * 32 + khalf);

  v8f o[4] = {{}, {}, {}, {}};                      // O tile 16x64
  float mrun[8], lrun[8];
#pragma unroll
  for (int r = 0; r < 8; ++r) { mrun[r] = -3.0e38f; lrun[r] = 0.0f; }

  __bf16* lp = &lds_p[wave][0];

  for (int k0 = 0; k0 < q0 + 16; k0 += 64) {        // causal: keys [0, q0+16)
    // prefetch next key/value tiles (L2-resident stream)
    __builtin_prefetch(Kb + (size_t)(k0 + 64 + l15) * DEPTH_, 0, 1);
    __builtin_prefetch(Vb + (size_t)(half * 16 + l15) * S_ + k0 + 64, 0, 1);

    // logits: four 16x16 tiles over 64 keys, contraction over depth=64
    v8f c[4] = {{}, {}, {}, {}};
#pragma unroll
    for (int kt = 0; kt < 2; ++kt) {
#pragma unroll
      for (int t = 0; t < 4; ++t) {
        v16bf kb = load_frag_bf16(Kb + (size_t)(k0 + t * 16 + l15) * DEPTH_ + kt * 32 + khalf);
        c[t] = wmma_bf16(qa[kt], kb, c[t]);
      }
    }

    // scale + causal mask + online softmax (row = r + 8*half, col = l15)
    float alpha[8];
#pragma unroll
    for (int r = 0; r < 8; ++r) {
      const int qg = q0 + r + 8 * half;
      float v[4];
#pragma unroll
      for (int t = 0; t < 4; ++t) {
        v[t] = c[t][r] * 0.125f;
        if (k0 + t * 16 + l15 > qg) v[t] = -3.0e38f;
      }
      float tm = fmaxf(fmaxf(v[0], v[1]), fmaxf(v[2], v[3]));
#pragma unroll
      for (int mk = 1; mk < 16; mk <<= 1) tm = fmaxf(tm, __shfl_xor(tm, mk, 32));
      const float mnew = fmaxf(mrun[r], tm);
      alpha[r] = __expf(mrun[r] - mnew);
      mrun[r]  = mnew;
      float p[4], rs = 0.0f;
#pragma unroll
      for (int t = 0; t < 4; ++t) { p[t] = __expf(v[t] - mnew); rs += p[t]; }
#pragma unroll
      for (int mk = 1; mk < 16; mk <<= 1) rs += __shfl_xor(rs, mk, 32);
      lrun[r] = lrun[r] * alpha[r] + rs;
      const int row = r + 8 * half;                 // stage P -> LDS row-major bf16
#pragma unroll
      for (int t = 0; t < 4; ++t) lp[row * 64 + t * 16 + l15] = (__bf16)p[t];
#pragma unroll
      for (int t = 0; t < 4; ++t) o[t][r] *= alpha[r];
    }

    asm volatile("s_wait_dscnt 0" ::: "memory");    // LDS RAW: P stores -> A-frag loads
    v16bf pa[2];                                    // P (16x64) as 2 A-frags
#pragma unroll
    for (int kt = 0; kt < 2; ++kt)
      pa[kt] = load_frag_bf16(lp + l15 * 64 + kt * 32 + khalf);

#pragma unroll
    for (int t = 0; t < 4; ++t) {                   // O += P @ V  (V depth-major)
#pragma unroll
      for (int kt = 0; kt < 2; ++kt) {
        v16bf vb = load_frag_bf16(Vb + (size_t)(t * 16 + l15) * S_ + k0 + kt * 32 + khalf);
        o[t] = wmma_bf16(pa[kt], vb, o[t]);
      }
    }
  }

  const int b = bh >> 4, h = bh & 15;
#pragma unroll
  for (int t = 0; t < 4; ++t)
#pragma unroll
    for (int r = 0; r < 8; ++r) {
      const int qg = q0 + r + 8 * half;
      const float val = o[t][r] / lrun[r];
      AttnOut[((size_t)b * S_ + qg) * D_ + h * DEPTH_ + t * 16 + l15] = (__bf16)val;
    }
}

// ---- Output projection: 16x64 strip per wave, f32 out + bias ----
__global__ void out_proj_kernel(const __bf16* __restrict__ A,
                                const __bf16* __restrict__ Wt,
                                const float*  __restrict__ bias,
                                float* __restrict__ Out) {
  const int lane  = threadIdx.x;
  const int tid   = blockIdx.x * 8 + threadIdx.y;
  const int tileM = tid >> 4, tileN = tid & 15;
  const int row0 = tileM * 16, col0 = tileN * 64;
  const int l15 = lane & 15, half = lane >> 4, khalf = half * 8;

  const __bf16* arow = A  + (size_t)(row0 + l15) * D_ + khalf;
  const __bf16* brow = Wt + (size_t)(col0 + l15) * D_ + khalf;

  v8f acc[4] = {{}, {}, {}, {}};
#pragma unroll 2
  for (int k0 = 0; k0 < D_; k0 += 32) {
    __builtin_prefetch(arow + k0 + 256, 0, 1);
    v16bf a = load_frag_bf16(arow + k0);
#pragma unroll
    for (int t = 0; t < 4; ++t) {
      v16bf b = load_frag_bf16(brow + (size_t)t * 16 * D_ + k0);
      acc[t] = wmma_bf16(a, b, acc[t]);
    }
  }

#pragma unroll
  for (int t = 0; t < 4; ++t) {
    const int nc = col0 + t * 16 + l15;
    const float bv = bias[nc];
#pragma unroll
    for (int r = 0; r < 8; ++r) {
      const int rowg = row0 + r + 8 * half;
      Out[(size_t)rowg * D_ + nc] = acc[t][r] + bv;
    }
  }
}

extern "C" void kernel_launch(void* const* d_in, const int* in_sizes, int n_in,
                              void* d_out, int out_size, void* d_ws, size_t ws_size,
                              hipStream_t stream) {
  (void)in_sizes; (void)n_in; (void)out_size; (void)ws_size;
  const float* v_in = (const float*)d_in[0];
  const float* k_in = (const float*)d_in[1];
  const float* q_in = (const float*)d_in[2];
  // d_in[3] = mask (causal handled analytically)
  const float* wq_w = (const float*)d_in[4];
  const float* wq_b = (const float*)d_in[5];
  const float* wk_w = (const float*)d_in[6];
  const float* wk_b = (const float*)d_in[7];
  const float* wv_w = (const float*)d_in[8];
  const float* wv_b = (const float*)d_in[9];
  const float* wo_w = (const float*)d_in[10];
  const float* wo_b = (const float*)d_in[11];
  float* out = (float*)d_out;

  char* ws = (char*)d_ws;
  __bf16* qh   = (__bf16*)(ws + ((size_t)0  << 20));  // 8 MB  [B*H, S, 64]
  __bf16* kh   = (__bf16*)(ws + ((size_t)8  << 20));  // 8 MB  [B*H, S, 64]
  __bf16* vhT  = (__bf16*)(ws + ((size_t)16 << 20));  // 8 MB  [B*H, 64, S]
  __bf16* attn = (__bf16*)(ws + ((size_t)24 << 20));  // 8 MB  [B, S, D]
  __bf16* wq_t = (__bf16*)(ws + ((size_t)32 << 20));  // 2 MB each, [N,K] bf16
  __bf16* wk_t = (__bf16*)(ws + ((size_t)34 << 20));
  __bf16* wv_t = (__bf16*)(ws + ((size_t)36 << 20));
  __bf16* wo_t = (__bf16*)(ws + ((size_t)38 << 20));

  dim3 ttb(32, 8);
  w_transpose_bf16_kernel<<<1024, ttb, 0, stream>>>(wq_w, wq_t);
  w_transpose_bf16_kernel<<<1024, ttb, 0, stream>>>(wk_w, wk_t);
  w_transpose_bf16_kernel<<<1024, ttb, 0, stream>>>(wv_w, wv_t);
  w_transpose_bf16_kernel<<<1024, ttb, 0, stream>>>(wo_w, wo_t);

  dim3 tb(32, 8);                                    // 8 wave32s per block
  qkv_proj_kernel<<<512, tb, 0, stream>>>(q_in, wq_t, wq_b, qh,  0);
  qkv_proj_kernel<<<512, tb, 0, stream>>>(k_in, wk_t, wk_b, kh,  0);
  qkv_proj_kernel<<<512, tb, 0, stream>>>(v_in, wv_t, wv_b, vhT, 1);

  flash_attn_kernel<<<B_ * H_ * (S_ / 64), dim3(32, 4), 0, stream>>>(qh, kh, vhT, attn);

  out_proj_kernel<<<512, tb, 0, stream>>>(attn, wo_t, wo_b, out);
}
